// PGMPNLayer_49787260895290
// MI455X (gfx1250) — matrix operand order repaired and checked by reading, not compiled, not run on hardware
//
#include <hip/hip_runtime.h>
#include <math.h>

using f16 = _Float16;
typedef __attribute__((ext_vector_type(8)))  f16   v8h;
typedef __attribute__((ext_vector_type(16))) f16   v16h;
typedef __attribute__((ext_vector_type(8)))  float v8f;

#define DEV __device__ __forceinline__

constexpr int Bc   = 16384;
constexpr int Dc   = 384;
constexpr int Hc   = 1024;
constexpr int DRc  = 512;
constexpr int DMc  = 1152;
constexpr int Sc   = 6;
constexpr int ROWS = Bc * Sc;     // 98304

// LDS staging: B tiles stored [128 rows][72 halves] (8-half pad -> conflict-free ds_load_b128)
constexpr int LDSB = 72;

// ---- gfx1250 async-to-LDS path (guarded; falls back to sync LDS staging) ----
#if defined(__has_builtin)
#if __has_builtin(__builtin_amdgcn_global_load_async_to_lds_b128)
#define HAVE_ASYNC_LDS 1
#endif
#endif
#ifndef HAVE_ASYNC_LDS
#define HAVE_ASYNC_LDS 0
#endif

#if HAVE_ASYNC_LDS
// exact parameter types per hipcc diagnostic: pointer to 16B int vector, AS1 src / AS3 dst
typedef int v4i_b128 __attribute__((vector_size(16)));
typedef __attribute__((address_space(1))) v4i_b128* as1_v4i;
typedef __attribute__((address_space(3))) v4i_b128* as3_v4i;
#if __has_builtin(__builtin_amdgcn_s_wait_asynccnt)
#define WAIT_ASYNC(n) __builtin_amdgcn_s_wait_asynccnt(n)
#else
#define WAIT_ASYNC(n) asm volatile("s_wait_asynccnt %0" ::"i"(n))
#endif
#else
#define WAIT_ASYNC(n)
#endif

// ---------------- helpers ----------------

DEV float wave_sum(float x) {
#pragma unroll
  for (int off = 16; off > 0; off >>= 1) x += __shfl_xor(x, off, 32);
  return x;
}

DEV float gelu_tanh(float x) {
  float x3 = x * x * x;
  return 0.5f * x * (1.0f + tanhf(0.7978845608028654f * (x + 0.044715f * x3)));
}

DEV float sigmoidf(float x) { return 1.0f / (1.0f + expf(-x)); }

// A fragment (16x32 f16): lane L<16 holds row, K = k0+{0..15};
// lane L>=16 holds K = k0+{16..31}.
DEV v16h frag_a(const f16* __restrict__ A, int lda, int row, int lane, int k0) {
  const f16* p = A + (size_t)row * lda + (k0 + ((lane >> 4) << 4));
  v8h lo = *(const v8h*)p;
  v8h hi = *(const v8h*)(p + 8);
  return __builtin_shufflevector(lo, hi, 0,1,2,3,4,5,6,7,8,9,10,11,12,13,14,15);
}

// B fragment from [N,K] row-major global (used by LRU kernel):
// lane L<16 -> K=k0..k0+15, lane L>=16 -> K=k0+16..k0+31.
DEV v16h frag_b(const f16* __restrict__ Bt, int ldb, int row, int lane, int k0) {
  const f16* p = Bt + (size_t)row * ldb + (k0 + ((lane >> 4) << 4));
  v8h lo = *(const v8h*)p;
  v8h hi = *(const v8h*)(p + 8);
  return __builtin_shufflevector(lo, hi, 0,1,2,3,4,5,6,7,8,9,10,11,12,13,14,15);
}

// B fragment from LDS-staged chunk tile ([128][LDSB] halves, kloc in {0,32})
DEV v16h frag_b_lds(const f16* sb, int row, int lane, int kloc) {
  const f16* p = sb + row * LDSB + (kloc + ((lane >> 4) << 4));
  v8h lo = *(const v8h*)p;
  v8h hi = *(const v8h*)(p + 8);
  return __builtin_shufflevector(lo, hi, 0,1,2,3,4,5,6,7,8,9,10,11,12,13,14,15);
}

DEV v8f wmma16(v16h a, v16h b, v8f c) {
  return __builtin_amdgcn_wmma_f32_16x16x32_f16(false, a, false, b, (short)0, c, false, false);
}

// Cooperatively stage one 128-row x 64-half chunk of Bt[N,K] into LDS.
// 256 threads x 4 b128 segments; async (ASYNCcnt) when available.
DEV void stage_b_chunk(const f16* __restrict__ Bt, int ldb, int nblock, int kc,
                       f16* sbuf, int tid) {
#pragma unroll
  for (int i = 0; i < 4; ++i) {
    const int seg = tid + i * 256;        // 0..1023
    const int row = seg >> 3;             // 0..127
    const int s8  = (seg & 7) << 3;       // 0..56 halves
    const f16* g = Bt + (size_t)(nblock + row) * ldb + kc + s8;
    f16* l = sbuf + row * LDSB + s8;
#if HAVE_ASYNC_LDS
    __builtin_amdgcn_global_load_async_to_lds_b128((as1_v4i)g, (as3_v4i)l, 0, 0);
#else
    *(v8h*)l = *(const v8h*)g;
#endif
  }
}

// ---------------- generic WMMA GEMM ----------------
// C[M,N] = A[M,K](f16) x Bt[N,K]^T(f16), f32 acc.  Block 64x128, 8 waves,
// wave = 32x32 (2x2 WMMA).  B double-buffered through LDS via async-to-LDS.

enum {
  EPI_F16 = 0,         // out f16 = acc
  EPI_BIAS_F16,        // out f16 = acc + bias[n]
  EPI_RELU_BIAS_F16,   // out f16 = relu(acc + bias[n])
  EPI_GELU_BIAS_F16,   // out f16 = gelu(acc + bias[n])
  EPI_NEGMLP_F32,      // out f32 = -(e0[idx]+e1[idx]) + acc + bias[n]
  EPI_RESID_F32        // out f32 += acc (+ bias[n] if bias)
};

template <int EPI>
__global__ __launch_bounds__(256)
void gemm_wmma(const f16* __restrict__ A, const f16* __restrict__ Bt,
               const float* __restrict__ bias, void* __restrict__ Out,
               const float* __restrict__ e0, const float* __restrict__ e1,
               int M, int N, int K) {
  __shared__ __align__(16) f16 sB[2][128 * LDSB];
  const int tid  = threadIdx.x;
  const int lane = tid & 31;
  const int w    = tid >> 5;
  const int m0 = blockIdx.y * 64 + (w >> 2) * 32;
  const int nblock = blockIdx.x * 128;
  const int n0 = nblock + (w & 3) * 32;

  v8f c00 = {}, c01 = {}, c10 = {}, c11 = {};
  const int ar0 = m0 + (lane & 15), ar1 = ar0 + 16;
  const int lbr0 = (w & 3) * 32 + (lane & 15), lbr1 = lbr0 + 16;

  const int nch = K >> 6;
  stage_b_chunk(Bt, K, nblock, 0, &sB[0][0], tid);
  for (int c = 0; c < nch; ++c) {
    if (c + 1 < nch) {
      stage_b_chunk(Bt, K, nblock, (c + 1) << 6, &sB[(c + 1) & 1][0], tid);
      WAIT_ASYNC(4);   // retire chunk c, leave chunk c+1 in flight
    } else {
      WAIT_ASYNC(0);
    }
    __syncthreads();
    const f16* bb = &sB[c & 1][0];
#pragma unroll
    for (int kk = 0; kk < 2; ++kk) {
      const int k0 = (c << 6) + kk * 32;
      if (k0 + 32 < K) {
        __builtin_prefetch(A + (size_t)ar0 * K + k0 + 32, 0, 1);
        __builtin_prefetch(A + (size_t)ar1 * K + k0 + 32, 0, 1);
      }
      v16h a0 = frag_a(A, K, ar0, lane, k0);
      v16h a1 = frag_a(A, K, ar1, lane, k0);
      v16h b0 = frag_b_lds(bb, lbr0, lane, kk * 32);
      v16h b1 = frag_b_lds(bb, lbr1, lane, kk * 32);
      c00 = wmma16(a0, b0, c00);
      c01 = wmma16(a0, b1, c01);
      c10 = wmma16(a1, b0, c10);
      c11 = wmma16(a1, b1, c11);
    }
    __syncthreads();  // protect buffer (c&1) from next-next issue
  }

  float* outf = (float*)Out;
  f16*   outh = (f16*)Out;
#pragma unroll
  for (int mt = 0; mt < 2; mt++) {
#pragma unroll
    for (int nt = 0; nt < 2; nt++) {
      v8f cc = mt ? (nt ? c11 : c10) : (nt ? c01 : c00);
      const int gn  = n0 + nt * 16 + (lane & 15);
      const int gmb = m0 + mt * 16 + ((lane >> 4) << 3);
      float bv = 0.0f;
      if constexpr (EPI == EPI_BIAS_F16 || EPI == EPI_RELU_BIAS_F16 ||
                    EPI == EPI_GELU_BIAS_F16 || EPI == EPI_NEGMLP_F32)
        bv = bias[gn];
      if constexpr (EPI == EPI_RESID_F32) { if (bias) bv = bias[gn]; }
#pragma unroll
      for (int q = 0; q < 8; q++) {
        size_t idx = (size_t)(gmb + q) * N + gn;
        float val = cc[q];
        if constexpr (EPI == EPI_F16)            outh[idx] = (f16)val;
        else if constexpr (EPI == EPI_BIAS_F16)  outh[idx] = (f16)(val + bv);
        else if constexpr (EPI == EPI_RELU_BIAS_F16) outh[idx] = (f16)fmaxf(val + bv, 0.0f);
        else if constexpr (EPI == EPI_GELU_BIAS_F16) outh[idx] = (f16)gelu_tanh(val + bv);
        else if constexpr (EPI == EPI_NEGMLP_F32) outf[idx] = val + bv - (e0[idx] + e1[idx]);
        else if constexpr (EPI == EPI_RESID_F32)  outf[idx] = outf[idx] + val + bv;
      }
    }
  }
}

// ---------------- dual gated GEMM: out f16 = gelu(A.Wg) * (A.Wv) ----------------

__global__ __launch_bounds__(256)
void gemm_dual_gated(const f16* __restrict__ A, const f16* __restrict__ Bg,
                     const f16* __restrict__ Bv, f16* __restrict__ Out,
                     int M, int N, int K) {
  __shared__ __align__(16) f16 sG[2][128 * LDSB];
  __shared__ __align__(16) f16 sV[2][128 * LDSB];
  const int tid  = threadIdx.x;
  const int lane = tid & 31;
  const int w    = tid >> 5;
  const int m0 = blockIdx.y * 64 + (w >> 2) * 32;
  const int nblock = blockIdx.x * 128;
  const int n0 = nblock + (w & 3) * 32;

  v8f g00 = {}, g01 = {}, g10 = {}, g11 = {};
  v8f v00 = {}, v01 = {}, v10 = {}, v11 = {};
  const int ar0 = m0 + (lane & 15), ar1 = ar0 + 16;
  const int lbr0 = (w & 3) * 32 + (lane & 15), lbr1 = lbr0 + 16;

  const int nch = K >> 6;
  stage_b_chunk(Bg, K, nblock, 0, &sG[0][0], tid);
  stage_b_chunk(Bv, K, nblock, 0, &sV[0][0], tid);
  for (int c = 0; c < nch; ++c) {
    if (c + 1 < nch) {
      stage_b_chunk(Bg, K, nblock, (c + 1) << 6, &sG[(c + 1) & 1][0], tid);
      stage_b_chunk(Bv, K, nblock, (c + 1) << 6, &sV[(c + 1) & 1][0], tid);
      WAIT_ASYNC(8);
    } else {
      WAIT_ASYNC(0);
    }
    __syncthreads();
    const f16* bgS = &sG[c & 1][0];
    const f16* bvS = &sV[c & 1][0];
#pragma unroll
    for (int kk = 0; kk < 2; ++kk) {
      const int k0 = (c << 6) + kk * 32;
      v16h a0  = frag_a(A, K, ar0, lane, k0);
      v16h a1  = frag_a(A, K, ar1, lane, k0);
      v16h bg0 = frag_b_lds(bgS, lbr0, lane, kk * 32);
      v16h bg1 = frag_b_lds(bgS, lbr1, lane, kk * 32);
      v16h bv0 = frag_b_lds(bvS, lbr0, lane, kk * 32);
      v16h bv1 = frag_b_lds(bvS, lbr1, lane, kk * 32);
      g00 = wmma16(a0, bg0, g00); g01 = wmma16(a0, bg1, g01);
      g10 = wmma16(a1, bg0, g10); g11 = wmma16(a1, bg1, g11);
      v00 = wmma16(a0, bv0, v00); v01 = wmma16(a0, bv1, v01);
      v10 = wmma16(a1, bv0, v10); v11 = wmma16(a1, bv1, v11);
    }
    __syncthreads();
  }

#pragma unroll
  for (int mt = 0; mt < 2; mt++) {
#pragma unroll
    for (int nt = 0; nt < 2; nt++) {
      v8f cg = mt ? (nt ? g11 : g10) : (nt ? g01 : g00);
      v8f cv = mt ? (nt ? v11 : v10) : (nt ? v01 : v00);
      const int gn  = n0 + nt * 16 + (lane & 15);
      const int gmb = m0 + mt * 16 + ((lane >> 4) << 3);
#pragma unroll
      for (int q = 0; q < 8; q++) {
        size_t idx = (size_t)(gmb + q) * N + gn;
        Out[idx] = (f16)(gelu_tanh(cg[q]) * cv[q]);
      }
    }
  }
}

// ---------------- fused RG-LRU (Griffin) ----------------
// WG: 16 batch rows x 512 channels; wave owns 64 channels (4 tiles).
// Recurrent state h lives in WMMA accumulator-layout registers across all 6 steps.

__global__ __launch_bounds__(256)
void rg_lru_fused(const f16* __restrict__ U, const f16* __restrict__ G,
                  const f16* __restrict__ Wrt, const f16* __restrict__ Wit,
                  const float* __restrict__ br, const float* __restrict__ bi,
                  const float* __restrict__ lam, f16* __restrict__ Z) {
  const int lane = threadIdx.x & 31;
  const int w    = threadIdx.x >> 5;
  const int m0 = blockIdx.x * 16;
  const int n0 = w * 64;

  float abase[4], rb[4], ib[4];
#pragma unroll
  for (int nt = 0; nt < 4; nt++) {
    int cn = n0 + nt * 16 + (lane & 15);
    float l  = lam[cn];
    float sp = (l > 20.0f) ? l : log1pf(expf(l));  // softplus
    abase[nt] = -8.0f * sp;                        // -RG_C * softplus(lam)
    rb[nt] = br[cn];
    ib[nt] = bi[cn];
  }

  v8f hst[4] = {};
  const int arow_b = (m0 + (lane & 15)) * Sc;
  const int brow   = n0 + (lane & 15);

  for (int s = 0; s < Sc; s++) {
    v8f cr[4] = {};
    v8f ci[4] = {};
    for (int k0 = 0; k0 < DRc; k0 += 32) {
      v16h a = frag_a(U, DRc, arow_b + s, lane, k0);
#pragma unroll
      for (int nt = 0; nt < 4; nt++) {
        v16h fr = frag_b(Wrt, DRc, brow + nt * 16, lane, k0);
        cr[nt] = wmma16(a, fr, cr[nt]);
        v16h fi = frag_b(Wit, DRc, brow + nt * 16, lane, k0);
        ci[nt] = wmma16(a, fi, ci[nt]);
      }
    }
#pragma unroll
    for (int nt = 0; nt < 4; nt++) {
      int cn = n0 + nt * 16 + (lane & 15);
#pragma unroll
      for (int q = 0; q < 8; q++) {
        int mrow = m0 + q + ((lane >> 4) << 3);
        size_t urow = (size_t)mrow * Sc + s;
        float r  = sigmoidf(cr[nt][q] + rb[nt]);
        float ii = sigmoidf(ci[nt][q] + ib[nt]);
        float a  = expf(abase[nt] * r);
        float xt = (float)U[urow * DRc + cn];
        float hv = hst[nt][q];
        hv = a * hv + sqrtf(fmaxf(0.0f, 1.0f - a * a)) * (ii * xt);
        hst[nt][q] = hv;
        float gv = (float)G[urow * DRc + cn];
        Z[urow * DRc + cn] = (f16)(hv * gv);
      }
    }
  }
}

// ---------------- small kernels ----------------

__global__ void transpose_w(const float* __restrict__ src, f16* __restrict__ dst,
                            int Kd, int Nd) {
  long idx = (long)blockIdx.x * 256 + threadIdx.x;
  if (idx >= (long)Kd * Nd) return;
  int n = (int)(idx / Kd);
  int k = (int)(idx - (long)n * Kd);
  dst[idx] = (f16)src[(size_t)k * Nd + n];
}

__global__ void make_mneg(const float* __restrict__ a, const float* __restrict__ b,
                          f16* __restrict__ dst, long n) {
  long idx = (long)blockIdx.x * 256 + threadIdx.x;
  if (idx >= n) return;
  dst[idx] = (f16)(a[idx] + b[idx]);
}

__global__ void cvt_f16(const float* __restrict__ src, f16* __restrict__ dst, long n) {
  long idx = (long)blockIdx.x * 256 + threadIdx.x;
  if (idx >= n) return;
  dst[idx] = (f16)src[idx];
}

__global__ __launch_bounds__(256)
void rmsnorm_rows(const float* __restrict__ x, const float* __restrict__ g,
                  f16* __restrict__ outp, int rows) {
  int r = (int)((blockIdx.x * blockDim.x + threadIdx.x) >> 5);
  if (r >= rows) return;
  int lane = threadIdx.x & 31;
  const float* xr = x + (size_t)r * Dc;
  float ss = 0.0f;
#pragma unroll
  for (int e = 0; e < 12; e++) { float v = xr[lane + e * 32]; ss += v * v; }
  ss = wave_sum(ss);
  float sc = rsqrtf(ss * (1.0f / Dc) + 1e-6f);
  f16* orow = outp + (size_t)r * Dc;
#pragma unroll
  for (int e = 0; e < 12; e++) { int c = lane + e * 32; orow[c] = (f16)(xr[c] * g[c] * sc); }
}

// attention-weight softmaxes + concat into x[B,6,D]
__global__ __launch_bounds__(256)
void build_x(const float* __restrict__ init_f, const float* __restrict__ head,
             const float* __restrict__ pred, const float* __restrict__ negf,
             const float* __restrict__ w_ap, const float* __restrict__ b_ap,
             const float* __restrict__ w_an, const float* __restrict__ b_an,
             float* __restrict__ X) {
  int b = (int)((blockIdx.x * blockDim.x + threadIdx.x) >> 5);
  if (b >= Bc) return;
  int lane = threadIdx.x & 31;
  float pv[3][12], nv[2][12], lg[3], ln[2];
#pragma unroll
  for (int p = 0; p < 3; p++) {
    float dot = 0.0f;
    const float* hp = head + ((size_t)p * Bc + b) * Dc;
    const float* pp = pred + ((size_t)p * Bc + b) * Dc;
#pragma unroll
    for (int e = 0; e < 12; e++) {
      int c = lane + e * 32;
      float m = hp[c] + pp[c];
      pv[p][e] = m;
      dot += m * w_ap[c];
    }
    lg[p] = wave_sum(dot) + b_ap[0];
  }
  float mx = fmaxf(lg[0], fmaxf(lg[1], lg[2]));
  float e0 = expf(lg[0] - mx), e1 = expf(lg[1] - mx), e2 = expf(lg[2] - mx);
  float inv = 1.0f / (e0 + e1 + e2);
  float awp[3] = {e0 * inv, e1 * inv, e2 * inv};
#pragma unroll
  for (int j = 0; j < 2; j++) {
    float dot = 0.0f;
    const float* np = negf + ((size_t)j * Bc + b) * Dc;
#pragma unroll
    for (int e = 0; e < 12; e++) {
      int c = lane + e * 32;
      float m = np[c];
      nv[j][e] = m;
      dot += m * w_an[c];
    }
    ln[j] = wave_sum(dot) + b_an[0];
  }
  float mn = fmaxf(ln[0], ln[1]);
  float f0 = expf(ln[0] - mn), f1 = expf(ln[1] - mn);
  float invn = 1.0f / (f0 + f1);
  float awn[2] = {f0 * invn, f1 * invn};
  float* xb = X + (size_t)b * Sc * Dc;
  const float* ib = init_f + (size_t)b * Dc;
#pragma unroll
  for (int e = 0; e < 12; e++) {
    int c = lane + e * 32;
    xb[c] = ib[c];
#pragma unroll
    for (int p = 0; p < 3; p++) xb[(1 + p) * Dc + c] = pv[p][e] * awp[p];
#pragma unroll
    for (int j = 0; j < 2; j++) xb[(4 + j) * Dc + c] = nv[j][e] * awn[j];
  }
}

// logits = fa.fa_wl + b, softmax over S, out = mean(x*w)
__global__ __launch_bounds__(256)
void finalize(const f16* __restrict__ FAb, const float* __restrict__ fa_wl,
              const float* __restrict__ fa_bl, const float* __restrict__ X,
              float* __restrict__ out) {
  int b = (int)((blockIdx.x * blockDim.x + threadIdx.x) >> 5);
  if (b >= Bc) return;
  int lane = threadIdx.x & 31;
  float lg[6];
#pragma unroll
  for (int s = 0; s < Sc; s++) {
    const f16* fr = FAb + ((size_t)b * Sc + s) * Dc;
    float dot = 0.0f;
#pragma unroll
    for (int e = 0; e < 12; e++) { int c = lane + e * 32; dot += (float)fr[c] * fa_wl[c]; }
    lg[s] = wave_sum(dot) + fa_bl[0];
  }
  float mx = lg[0];
#pragma unroll
  for (int s = 1; s < Sc; s++) mx = fmaxf(mx, lg[s]);
  float wsum = 0.0f, wv[6];
#pragma unroll
  for (int s = 0; s < Sc; s++) { wv[s] = expf(lg[s] - mx); wsum += wv[s]; }
  float invw = 1.0f / wsum;
  const float* xb = X + (size_t)b * Sc * Dc;
  float* ob = out + (size_t)b * Dc;
#pragma unroll
  for (int e = 0; e < 12; e++) {
    int c = lane + e * 32;
    float acc = 0.0f;
#pragma unroll
    for (int s = 0; s < Sc; s++) acc += xb[s * Dc + c] * wv[s];
    ob[c] = acc * invw * (1.0f / 6.0f);
  }
}

// ---------------- host launcher ----------------

extern "C" void kernel_launch(void* const* d_in, const int* in_sizes, int n_in,
                              void* d_out, int out_size, void* d_ws, size_t ws_size,
                              hipStream_t stream) {
  const float* init_f = (const float*)d_in[0];
  const float* head   = (const float*)d_in[1];
  const float* pred   = (const float*)d_in[2];
  const float* w_ap = (const float*)d_in[3];  const float* b_ap = (const float*)d_in[4];
  const float* w_an = (const float*)d_in[5];  const float* b_an = (const float*)d_in[6];
  const float* nt_w1 = (const float*)d_in[7]; const float* nt_b1 = (const float*)d_in[8];
  const float* nt_w2 = (const float*)d_in[9]; const float* nt_b2 = (const float*)d_in[10];
  const float* g1 = (const float*)d_in[11];
  const float* rg_wx = (const float*)d_in[12]; const float* rg_bx = (const float*)d_in[13];
  const float* rg_wy = (const float*)d_in[14]; const float* rg_by = (const float*)d_in[15];
  const float* rg_wr = (const float*)d_in[16]; const float* rg_br = (const float*)d_in[17];
  const float* rg_wi = (const float*)d_in[18]; const float* rg_bi = (const float*)d_in[19];
  const float* rg_lam = (const float*)d_in[20];
  const float* rg_wo = (const float*)d_in[21]; const float* rg_bo = (const float*)d_in[22];
  const float* g2 = (const float*)d_in[23];
  const float* mlp_wg = (const float*)d_in[24]; const float* mlp_wv = (const float*)d_in[25];
  const float* mlp_wo = (const float*)d_in[26];
  const float* fa_wg = (const float*)d_in[27]; const float* fa_wv = (const float*)d_in[28];
  const float* fa_wo = (const float*)d_in[29];
  const float* fa_wl = (const float*)d_in[30]; const float* fa_bl = (const float*)d_in[31];
  float* out = (float*)d_out;

  char* ws = (char*)d_ws;
  size_t off = 0;
  auto takep = [&](size_t bytes) -> char* {
    char* r = ws + off;
    off = (off + bytes + 255) & ~(size_t)255;
    return r;
  };

  // f16 transposed weights ([N,K] layout), ~7.4 MB
  f16* nt_w1t = (f16*)takep((size_t)Hc * Dc * 2);
  f16* nt_w2t = (f16*)takep((size_t)Dc * Hc * 2);
  f16* rg_wxt = (f16*)takep((size_t)DRc * Dc * 2);
  f16* rg_wyt = (f16*)takep((size_t)DRc * Dc * 2);
  f16* rg_wrt = (f16*)takep((size_t)DRc * DRc * 2);
  f16* rg_wit = (f16*)takep((size_t)DRc * DRc * 2);
  f16* rg_wot = (f16*)takep((size_t)Dc * DRc * 2);
  f16* mlp_wgt = (f16*)takep((size_t)DMc * Dc * 2);
  f16* mlp_wvt = (f16*)takep((size_t)DMc * Dc * 2);
  f16* mlp_wot = (f16*)takep((size_t)Dc * DMc * 2);
  f16* fa_wgt = (f16*)takep((size_t)Dc * Dc * 2);
  f16* fa_wvt = (f16*)takep((size_t)Dc * Dc * 2);
  f16* fa_wot = (f16*)takep((size_t)Dc * Dc * 2);

  float* X  = (float*)takep((size_t)ROWS * Dc * 4);  // 151 MB, persistent
  f16*   Hh = (f16*)takep((size_t)ROWS * Dc * 2);    // 75.5 MB, normed/f16 activations

  // phase-aliased pool (peak 302 MB)
  char* pool = takep((size_t)301989888);
  f16*   mneg = (f16*)(pool);                                  // [2B,D]  phase 1
  f16*   hid  = (f16*)(pool + (size_t)25165824);               // [2B,H]  phase 1
  float* negf = (float*)(pool + (size_t)92274688);             // [2B,D]  phase 1-2
  f16*   U    = (f16*)(pool);                                  // [R,DR]  phase 3
  f16*   Gt   = (f16*)(pool + (size_t)100663296);              // [R,DR]  phase 3
  f16*   Zb   = (f16*)(pool + (size_t)201326592);              // [R,DR]  phase 3
  f16*   MP   = (f16*)(pool);                                  // [R,DM]  phase 4
  f16*   FG   = (f16*)(pool);                                  // [R,D]   phase 5
  f16*   FAb  = (f16*)(pool + (size_t)100663296);              // [R,D]   phase 5
  if (ws_size < off) return;  // insufficient scratch: bail deterministically

  // ---- phase 0: weight convert+transpose (tiny, L2-resident) ----
  auto tr = [&](const float* src, f16* dst, int Kd, int Nd) {
    int n = Kd * Nd;
    transpose_w<<<(n + 255) / 256, 256, 0, stream>>>(src, dst, Kd, Nd);
  };
  tr(nt_w1, nt_w1t, Dc, Hc);   tr(nt_w2, nt_w2t, Hc, Dc);
  tr(rg_wx, rg_wxt, Dc, DRc);  tr(rg_wy, rg_wyt, Dc, DRc);
  tr(rg_wr, rg_wrt, DRc, DRc); tr(rg_wi, rg_wit, DRc, DRc);
  tr(rg_wo, rg_wot, DRc, Dc);
  tr(mlp_wg, mlp_wgt, Dc, DMc); tr(mlp_wv, mlp_wvt, Dc, DMc);
  tr(mlp_wo, mlp_wot, DMc, Dc);
  tr(fa_wg, fa_wgt, Dc, Dc); tr(fa_wv, fa_wvt, Dc, Dc); tr(fa_wo, fa_wot, Dc, Dc);

  // ---- phase 1: negated-atomic residual MLP ----
  {
    long n = (long)2 * Bc * Dc;
    make_mneg<<<(n + 255) / 256, 256, 0, stream>>>(
        head + (size_t)3 * Bc * Dc, pred + (size_t)3 * Bc * Dc, mneg, n);
  }
  gemm_wmma<EPI_RELU_BIAS_F16><<<dim3(Hc / 128, (2 * Bc) / 64), 256, 0, stream>>>(
      mneg, nt_w1t, nt_b1, hid, nullptr, nullptr, 2 * Bc, Hc, Dc);
  gemm_wmma<EPI_NEGMLP_F32><<<dim3(Dc / 128, (2 * Bc) / 64), 256, 0, stream>>>(
      hid, nt_w2t, nt_b2, negf, head + (size_t)3 * Bc * Dc, pred + (size_t)3 * Bc * Dc,
      2 * Bc, Dc, Hc);

  // ---- phase 2: grouped softmax attention + concat -> x[B,6,D] ----
  build_x<<<Bc / 8, 256, 0, stream>>>(init_f, head, pred, negf, w_ap, b_ap, w_an, b_an, X);

  // ---- phase 3: Griffin recurrent branch ----
  rmsnorm_rows<<<ROWS / 8, 256, 0, stream>>>(X, g1, Hh, ROWS);
  gemm_wmma<EPI_BIAS_F16><<<dim3(DRc / 128, ROWS / 64), 256, 0, stream>>>(
      Hh, rg_wxt, rg_bx, U, nullptr, nullptr, ROWS, DRc, Dc);
  gemm_wmma<EPI_GELU_BIAS_F16><<<dim3(DRc / 128, ROWS / 64), 256, 0, stream>>>(
      Hh, rg_wyt, rg_by, Gt, nullptr, nullptr, ROWS, DRc, Dc);
  rg_lru_fused<<<Bc / 16, 256, 0, stream>>>(U, Gt, rg_wrt, rg_wit, rg_br, rg_bi, rg_lam, Zb);
  gemm_wmma<EPI_RESID_F32><<<dim3(Dc / 128, ROWS / 64), 256, 0, stream>>>(
      Zb, rg_wot, rg_bo, X, nullptr, nullptr, ROWS, Dc, DRc);

  // ---- phase 4: gated-MLP residual branch ----
  rmsnorm_rows<<<ROWS / 8, 256, 0, stream>>>(X, g2, Hh, ROWS);
  gemm_dual_gated<<<dim3(DMc / 128, ROWS / 64), 256, 0, stream>>>(
      Hh, mlp_wgt, mlp_wvt, MP, ROWS, DMc, Dc);
  gemm_wmma<EPI_RESID_F32><<<dim3(Dc / 128, ROWS / 64), 256, 0, stream>>>(
      MP, mlp_wot, nullptr, X, nullptr, nullptr, ROWS, Dc, DMc);

  // ---- phase 5: final attention aggregation ----
  {
    long n = (long)ROWS * Dc;
    cvt_f16<<<(n + 255) / 256, 256, 0, stream>>>(X, Hh, n);
  }
  gemm_dual_gated<<<dim3(Dc / 128, ROWS / 64), 256, 0, stream>>>(
      Hh, fa_wgt, fa_wvt, FG, ROWS, Dc, Dc);
  gemm_wmma<EPI_F16><<<dim3(Dc / 128, ROWS / 64), 256, 0, stream>>>(
      FG, fa_wot, nullptr, FAb, nullptr, nullptr, ROWS, Dc, Dc);
  finalize<<<Bc / 8, 256, 0, stream>>>(FAb, fa_wl, fa_bl, X, out);
}
